// PropagationNetwork_43860206026999
// MI455X (gfx1250) — matrix operand order repaired and checked by reading, not compiled, not run on hardware
//
#include <hip/hip_runtime.h>

#define B_   32
#define R_   4096
#define N_   64
#define H_   128
#define MREL (B_*R_)   // 131072 relation rows
#define MOBJ (B_*N_)   // 2048 object rows
#define KROWS 64       // K-chunk (relation rows) per effect iteration
#define BSTRIDE 130    // padded LDS row stride in halfs (128 + 1 DWORD pad)

typedef __attribute__((ext_vector_type(16))) _Float16 v16h;
typedef __attribute__((ext_vector_type(8)))  float    v8f;
typedef __attribute__((ext_vector_type(4)))  float    f32x4;

__device__ __forceinline__ v8f zero8() {
  v8f z = {0.f,0.f,0.f,0.f,0.f,0.f,0.f,0.f};
  return z;
}

__device__ __forceinline__ v8f wmma_f16(v16h a, v16h b, v8f c) {
  // D = A(16x32 f16) * B(32x16 f16) + C(16x16 f32)
  return __builtin_amdgcn_wmma_f32_16x16x32_f16(false, a, false, b, (short)0, c, false, false);
}

// A fragment (16x32 f16) from row-major f16 source (row stride in halfs).
__device__ __forceinline__ v16h afrag_h(const _Float16* base, int stride) {
  const int lane = threadIdx.x & 31;
  const int m = lane & 15, kb = (lane >> 4) * 8;
  const _Float16* p = base + m * stride + kb;
  v16h a;
  *reinterpret_cast<f32x4*>(&a)       = *reinterpret_cast<const f32x4*>(p);
  *(reinterpret_cast<f32x4*>(&a) + 1) = *reinterpret_cast<const f32x4*>(p + 16);
  return a;
}

// A fragment from row-major f32 source (convert on the fly).
__device__ __forceinline__ v16h afrag_f32(const float* base, int stride) {
  const int lane = threadIdx.x & 31;
  const int m = lane & 15, kb = (lane >> 4) * 8;
  const float* p = base + m * stride + kb;
  v16h a;
#pragma unroll
  for (int i = 0; i < 8; ++i) { a[i] = (_Float16)p[i]; a[i + 8] = (_Float16)p[i + 16]; }
  return a;
}

// B fragment (32x16 f16) from an N-major ("B^T", [n][k]) f16 buffer.
__device__ __forceinline__ v16h bfrag(const _Float16* bt, int stride) {
  const int lane = threadIdx.x & 31;
  const int n = lane & 15, ks = (lane >> 4) * 8;
  const _Float16* p = bt + n * stride + ks;
  v16h b;
  *reinterpret_cast<f32x4*>(&b)       = *reinterpret_cast<const f32x4*>(p);
  *(reinterpret_cast<f32x4*>(&b) + 1) = *reinterpret_cast<const f32x4*>(p + 16);
  return b;
}

// B fragment from a K-major (row-major [k][n]) f16 buffer: strided column reads.
__device__ __forceinline__ v16h bfrag_km(const _Float16* b, int stride) {
  const int lane = threadIdx.x & 31;
  const int n = lane & 15, ks = (lane >> 4) * 8;
  v16h r;
#pragma unroll
  for (int i = 0; i < 16; ++i) r[i] = b[(ks + (i < 8 ? i : i + 8)) * stride + n];
  return r;
}

// ---------------------------------------------------------------------------
// TDM support (Tensor Data Mover): DMA f16 tile (rows x 128) global -> LDS,
// with 1-DWORD pad after every 64 DWORDs (=> LDS row stride 130 halfs).
// ---------------------------------------------------------------------------
#if defined(__AMDGCN__) && __has_builtin(__builtin_amdgcn_tensor_load_to_lds)
#define HAVE_TDM 1
#if __has_include(<hip/amd_detail/amd_gfx1250_TDM.h>)
#define TDM_SIX_ARG 1
#endif
typedef unsigned int v4u __attribute__((ext_vector_type(4)));
typedef int          v4i __attribute__((ext_vector_type(4)));
typedef int          v8i __attribute__((ext_vector_type(8)));

__device__ __forceinline__ void tdm_load_tile_f16(const _Float16* gsrc, _Float16* ldst,
                                                  unsigned rows) {
  const unsigned long long ga = (unsigned long long)(uintptr_t)gsrc;
  const unsigned lds = (unsigned)(uintptr_t)ldst;   // low 32 bits = LDS byte offset
  v4u g0;
  g0[0] = 1u;                                     // count=1 (valid user descriptor)
  g0[1] = lds;                                    // lds_addr (bytes)
  g0[2] = (unsigned)ga;                           // global_addr[31:0]
  g0[3] = (unsigned)(ga >> 32) | (2u << 30);      // global_addr[56:32] | type=2
  v8i g1;
  // data_size=2B (1<<16) | pad_enable (1<<20) | pad_interval=64 DWORDs (5<<22) | pad_amount=1 DWORD (0<<25)
  g1[0] = (int)0x01510000u;
  g1[1] = (int)(128u << 16);                      // tensor_dim0 = 128 (bits 79:48, low half)
  g1[2] = (int)(rows << 16);                      // tensor_dim1 = rows (bits 111:80, low half)
  g1[3] = (int)(128u << 16);                      // tile_dim0 = 128 (bits 127:112)
  g1[4] = (int)rows;                              // tile_dim1 = rows (bits 143:128)
  g1[5] = 128;                                    // tensor_dim0_stride = 128 elements
  g1[6] = 0;
  g1[7] = 0;
  v4i z4 = {0, 0, 0, 0};
#if defined(TDM_SIX_ARG)
  v8i z8 = {0, 0, 0, 0, 0, 0, 0, 0};
  __builtin_amdgcn_tensor_load_to_lds(g0, g1, z4, z4, z8, 0);
#else
  __builtin_amdgcn_tensor_load_to_lds(g0, g1, z4, z4, 0);
#endif
}
#endif

// ---------------------------------------------------------------------------
// Prep: f16-transpose all weights into B^T layout; zero prop state buffers.
// ---------------------------------------------------------------------------
__global__ void prep_kernel(const float* rm_w2, const float* rm_w3, const float* om_w2,
                            const float* ov_w1, const float* rmp_w, const float* omp_w,
                            _Float16* w2T, _Float16* w3T, _Float16* omw2T, _Float16* ovw1T,
                            _Float16* rmpT, _Float16* ompT,
                            float* propF, _Float16* propH, _Float16* propT) {
  const int idx = blockIdx.x * blockDim.x + threadIdx.x;  // grid covers MOBJ*H_
  if (idx < H_ * H_) {
    const int k = idx >> 7, n = idx & 127;   // W is [K][N] row-major
    w2T  [n * H_ + k] = (_Float16)rm_w2[idx];
    w3T  [n * H_ + k] = (_Float16)rm_w3[idx];
    omw2T[n * H_ + k] = (_Float16)om_w2[idx];
    ovw1T[n * H_ + k] = (_Float16)ov_w1[idx];
  }
  if (idx < 3 * H_ * H_) {
    const int k = idx >> 7, n = idx & 127;   // [384][128] -> [128][384]
    rmpT[n * (3 * H_) + k] = (_Float16)rmp_w[idx];
    ompT[n * (3 * H_) + k] = (_Float16)omp_w[idx];
  }
  if (idx < MOBJ * H_) {
    propF[idx] = 0.f;
    propH[idx] = (_Float16)0.f;
    propT[idx] = (_Float16)0.f;
  }
}

// ---------------------------------------------------------------------------
// rel_enc = relu(relu(relu(rel@w1+b1)@w2+b2)@w3+b3), 16 rows per wave.
// ---------------------------------------------------------------------------
__global__ void __launch_bounds__(32) encode_rel_kernel(
    const float* __restrict__ rel, const float* __restrict__ w1, const float* __restrict__ b1,
    const _Float16* __restrict__ w2T, const float* __restrict__ b2,
    const _Float16* __restrict__ w3T, const float* __restrict__ b3,
    _Float16* __restrict__ relEnc) {
  __shared__ _Float16 hbuf[16 * H_];
  __shared__ float rrow[16 * 4];
  const int blk = blockIdx.x, lane = threadIdx.x;
  const float* rbase = rel + (size_t)blk * 16 * 4;
  for (int i = lane; i < 64; i += 32) rrow[i] = rbase[i];
  __syncthreads();
  // layer 1 (K=4, scalar VALU)
#pragma unroll
  for (int t = 0; t < 4; ++t) {
    const int c = t * 32 + lane;
    const float wa = w1[c], wb = w1[H_ + c], wc = w1[2 * H_ + c], wd = w1[3 * H_ + c];
    const float bb = b1[c];
#pragma unroll
    for (int m = 0; m < 16; ++m) {
      float v = bb + rrow[m*4+0]*wa + rrow[m*4+1]*wb + rrow[m*4+2]*wc + rrow[m*4+3]*wd;
      hbuf[m * H_ + c] = (_Float16)fmaxf(v, 0.f);
    }
  }
  __syncthreads();

  const int n = lane & 15, mh = (lane >> 4) * 8;
  v8f acc[8];
  // layer 2 (WMMA, K=128)
#pragma unroll
  for (int t = 0; t < 8; ++t) acc[t] = zero8();
#pragma unroll
  for (int kc = 0; kc < 4; ++kc) {
    v16h a = afrag_h(hbuf + kc * 32, H_);
#pragma unroll
    for (int t = 0; t < 8; ++t)
      acc[t] = wmma_f16(a, bfrag(w2T + t * 16 * H_ + kc * 32, H_), acc[t]);
  }
  __syncthreads();
#pragma unroll
  for (int t = 0; t < 8; ++t) {
    const float bb = b2[t * 16 + n];
#pragma unroll
    for (int j = 0; j < 8; ++j)
      hbuf[(mh + j) * H_ + t * 16 + n] = (_Float16)fmaxf(acc[t][j] + bb, 0.f);
  }
  __syncthreads();
  // layer 3 (WMMA, K=128) -> global f16
#pragma unroll
  for (int t = 0; t < 8; ++t) acc[t] = zero8();
#pragma unroll
  for (int kc = 0; kc < 4; ++kc) {
    v16h a = afrag_h(hbuf + kc * 32, H_);
#pragma unroll
    for (int t = 0; t < 8; ++t)
      acc[t] = wmma_f16(a, bfrag(w3T + t * 16 * H_ + kc * 32, H_), acc[t]);
  }
  _Float16* orow = relEnc + (size_t)blk * 16 * H_;
#pragma unroll
  for (int t = 0; t < 8; ++t) {
    const float bb = b3[t * 16 + n];
#pragma unroll
    for (int j = 0; j < 8; ++j)
      orow[(mh + j) * H_ + t * 16 + n] = (_Float16)fmaxf(acc[t][j] + bb, 0.f);
  }
}

// ---------------------------------------------------------------------------
// obj_enc = relu(relu(obj@w1+b1)@w2+b2)
// ---------------------------------------------------------------------------
__global__ void __launch_bounds__(32) encode_obj_kernel(
    const float* __restrict__ obj, const float* __restrict__ w1, const float* __restrict__ b1,
    const _Float16* __restrict__ w2T, const float* __restrict__ b2,
    _Float16* __restrict__ objEnc) {
  __shared__ _Float16 hbuf[16 * H_];
  __shared__ float orow[16 * 6];
  const int blk = blockIdx.x, lane = threadIdx.x;
  const float* rbase = obj + (size_t)blk * 16 * 6;
  for (int i = lane; i < 96; i += 32) orow[i] = rbase[i];
  __syncthreads();
#pragma unroll
  for (int t = 0; t < 4; ++t) {
    const int c = t * 32 + lane;
    float wk[6];
#pragma unroll
    for (int k = 0; k < 6; ++k) wk[k] = w1[k * H_ + c];
    const float bb = b1[c];
#pragma unroll
    for (int m = 0; m < 16; ++m) {
      float v = bb;
#pragma unroll
      for (int k = 0; k < 6; ++k) v += orow[m * 6 + k] * wk[k];
      hbuf[m * H_ + c] = (_Float16)fmaxf(v, 0.f);
    }
  }
  __syncthreads();
  const int n = lane & 15, mh = (lane >> 4) * 8;
  v8f acc[8];
#pragma unroll
  for (int t = 0; t < 8; ++t) acc[t] = zero8();
#pragma unroll
  for (int kc = 0; kc < 4; ++kc) {
    v16h a = afrag_h(hbuf + kc * 32, H_);
#pragma unroll
    for (int t = 0; t < 8; ++t)
      acc[t] = wmma_f16(a, bfrag(w2T + t * 16 * H_ + kc * 32, H_), acc[t]);
  }
  _Float16* orow2 = objEnc + (size_t)blk * 16 * H_;
#pragma unroll
  for (int t = 0; t < 8; ++t) {
    const float bb = b2[t * 16 + n];
#pragma unroll
    for (int j = 0; j < 8; ++j)
      orow2[(mh + j) * H_ + t * 16 + n] = (_Float16)fmaxf(acc[t][j] + bb, 0.f);
  }
}

// ---------------------------------------------------------------------------
// Per prop step, stage 1: sp = S@prop, rp = Rv@prop (K=64), then
// rel_out = relu([rel_enc|sp|rp] @ rmp_w + b) as 12 K-chunks from 3 sources.
// ---------------------------------------------------------------------------
__global__ void __launch_bounds__(32) rel_step_kernel(
    const float* __restrict__ sender, const float* __restrict__ receiver,
    const _Float16* __restrict__ relEnc, const _Float16* __restrict__ propT,
    const _Float16* __restrict__ rmpT, const float* __restrict__ rmp_b,
    _Float16* __restrict__ relOut) {
  __shared__ _Float16 sp[16 * H_];
  __shared__ _Float16 rp[16 * H_];
  const int blk = blockIdx.x, lane = threadIdx.x;
  const int b = blk >> 8;                        // 256 16-row tiles per batch
  const _Float16* pT = propT + (size_t)b * H_ * N_;   // [h][obj] f16
  const int n = lane & 15, mh = (lane >> 4) * 8;

  v8f acc[8];
  // senders_prop = S(16x64) @ prop(64x128)
#pragma unroll
  for (int t = 0; t < 8; ++t) acc[t] = zero8();
  const float* sbase = sender + (size_t)blk * 16 * N_;
#pragma unroll
  for (int kc = 0; kc < 2; ++kc) {
    v16h a = afrag_f32(sbase + kc * 32, N_);
#pragma unroll
    for (int t = 0; t < 8; ++t)
      acc[t] = wmma_f16(a, bfrag(pT + t * 16 * N_ + kc * 32, N_), acc[t]);
  }
#pragma unroll
  for (int t = 0; t < 8; ++t)
#pragma unroll
    for (int j = 0; j < 8; ++j)
      sp[(mh + j) * H_ + t * 16 + n] = (_Float16)acc[t][j];

  // receivers_prop
#pragma unroll
  for (int t = 0; t < 8; ++t) acc[t] = zero8();
  const float* rbase = receiver + (size_t)blk * 16 * N_;
#pragma unroll
  for (int kc = 0; kc < 2; ++kc) {
    v16h a = afrag_f32(rbase + kc * 32, N_);
#pragma unroll
    for (int t = 0; t < 8; ++t)
      acc[t] = wmma_f16(a, bfrag(pT + t * 16 * N_ + kc * 32, N_), acc[t]);
  }
#pragma unroll
  for (int t = 0; t < 8; ++t)
#pragma unroll
    for (int j = 0; j < 8; ++j)
      rp[(mh + j) * H_ + t * 16 + n] = (_Float16)acc[t][j];
  __syncthreads();

  // rel_out: K = 384 = [rel_enc(128) | sp(128) | rp(128)]
#pragma unroll
  for (int t = 0; t < 8; ++t) acc[t] = zero8();
  const _Float16* ebase = relEnc + (size_t)blk * 16 * H_;
#pragma unroll
  for (int kc = 0; kc < 12; ++kc) {
    v16h a;
    if (kc < 4)      a = afrag_h(ebase + kc * 32, H_);
    else if (kc < 8) a = afrag_h(sp + (kc - 4) * 32, H_);
    else             a = afrag_h(rp + (kc - 8) * 32, H_);
#pragma unroll
    for (int t = 0; t < 8; ++t)
      acc[t] = wmma_f16(a, bfrag(rmpT + t * 16 * (3 * H_) + kc * 32, 3 * H_), acc[t]);
  }
  _Float16* orow = relOut + (size_t)blk * 16 * H_;
#pragma unroll
  for (int t = 0; t < 8; ++t) {
    const float bb = rmp_b[t * 16 + n];
#pragma unroll
    for (int j = 0; j < 8; ++j)
      orow[(mh + j) * H_ + t * 16 + n] = (_Float16)fmaxf(acc[t][j] + bb, 0.f);
  }
}

// ---------------------------------------------------------------------------
// effect[b,n,h] = sum_r recv[b,r,n] * rel_out[b,r,h]  -- 64x4096x128 per batch.
// 8 waves/block; each wave owns one 16-wide h tile. rel_out K-chunks are DMA'd
// into LDS by the Tensor Data Mover (async, pad-striped rows, s_wait_tensorcnt);
// recv chunk is f32->f16 converted by the waves while the TDM runs.
// ---------------------------------------------------------------------------
__global__ void __launch_bounds__(256) effect_kernel(
    const float* __restrict__ receiver, const _Float16* __restrict__ relOut,
    float* __restrict__ effect) {
  __shared__ _Float16 ldsA[16 * KROWS];       // recv^T chunk: [n][k]
  __shared__ _Float16 ldsB[KROWS * BSTRIDE];  // rel_out chunk: [k][h] (padded rows)
  const int blk = blockIdx.x;
  const int b = blk >> 2, nt = blk & 3;
  const int tid = threadIdx.x, wave = tid >> 5, lane = tid & 31;
  v8f acc = zero8();
  for (int kc = 0; kc < R_ / KROWS; ++kc) {
    const int k0 = kc * KROWS;
    __syncthreads();   // previous chunk's WMMAs done before LDS is overwritten
#if defined(HAVE_TDM)
    if (wave == 0)     // one wave issues the DMA for the whole block
      tdm_load_tile_f16(relOut + ((size_t)b * R_ + k0) * H_, ldsB, KROWS);
#else
    for (int i = tid; i < KROWS * H_; i += 256) {
      const int k = i >> 7, h2 = i & 127;
      ldsB[k * BSTRIDE + h2] = relOut[((size_t)b * R_ + k0 + k) * H_ + h2];
    }
#endif
    // stage recv^T (f32 -> f16) while the tensor DMA is in flight
    for (int i = tid; i < 16 * KROWS; i += 256) {
      const int k = i >> 4, n2 = i & 15;
      ldsA[n2 * KROWS + k] =
          (_Float16)receiver[((size_t)b * R_ + k0 + k) * N_ + nt * 16 + n2];
    }
#if defined(HAVE_TDM) && __has_builtin(__builtin_amdgcn_s_wait_tensorcnt)
    if (wave == 0) __builtin_amdgcn_s_wait_tensorcnt(0);
#endif
    __syncthreads();
#pragma unroll
    for (int kk = 0; kk < KROWS / 32; ++kk) {
      v16h a  = afrag_h(ldsA + kk * 32, KROWS);
      v16h bf = bfrag_km(ldsB + (size_t)kk * 32 * BSTRIDE + wave * 16, BSTRIDE);
      acc = wmma_f16(a, bf, acc);
    }
  }
  const int n = lane & 15, mh = (lane >> 4) * 8;
#pragma unroll
  for (int j = 0; j < 8; ++j)
    effect[((size_t)b * N_ + nt * 16 + mh + j) * H_ + wave * 16 + n] = acc[j];
}

// ---------------------------------------------------------------------------
// prop = relu([obj_enc|effect|prop] @ omp_w + b + prop); writes prop in
// f32 + f16 row-major and f16 transposed ([b][h][obj]) for the next rel_step.
// ---------------------------------------------------------------------------
__global__ void __launch_bounds__(32) obj_step_kernel(
    const _Float16* __restrict__ objEnc, const float* __restrict__ effect,
    float* __restrict__ propF, _Float16* __restrict__ propH, _Float16* __restrict__ propT,
    const _Float16* __restrict__ ompT, const float* __restrict__ omp_b) {
  const int blk = blockIdx.x, lane = threadIdx.x;
  const size_t row0 = (size_t)blk * 16;
  v8f acc[8];
#pragma unroll
  for (int t = 0; t < 8; ++t) acc[t] = zero8();
#pragma unroll
  for (int kc = 0; kc < 12; ++kc) {
    v16h a;
    if (kc < 4)      a = afrag_h(objEnc + row0 * H_ + kc * 32, H_);
    else if (kc < 8) a = afrag_f32(effect + row0 * H_ + (kc - 4) * 32, H_);
    else             a = afrag_h(propH + row0 * H_ + (kc - 8) * 32, H_);
#pragma unroll
    for (int t = 0; t < 8; ++t)
      acc[t] = wmma_f16(a, bfrag(ompT + t * 16 * (3 * H_) + kc * 32, 3 * H_), acc[t]);
  }
  const int b = blk >> 2, objbase = (blk & 3) * 16;
  const int n = lane & 15, mh = (lane >> 4) * 8;
#pragma unroll
  for (int t = 0; t < 8; ++t) {
    const int col = t * 16 + n;
    const float bb = omp_b[col];
#pragma unroll
    for (int j = 0; j < 8; ++j) {
      const size_t row = row0 + mh + j;
      float v = fmaxf(acc[t][j] + bb + propF[row * H_ + col], 0.f);
      propF[row * H_ + col] = v;
      const _Float16 hv = (_Float16)v;
      propH[row * H_ + col] = hv;
      propT[((size_t)b * H_ + col) * N_ + objbase + mh + j] = hv;
    }
  }
}

// ---------------------------------------------------------------------------
// out = relu(prop@ov_w1+b1) @ ov_w2 + b2   (final head, OUT=3)
// ---------------------------------------------------------------------------
__global__ void __launch_bounds__(32) out_head_kernel(
    const _Float16* __restrict__ propH, const _Float16* __restrict__ ovw1T,
    const float* __restrict__ ov_b1, const float* __restrict__ ov_w2,
    const float* __restrict__ ov_b2, float* __restrict__ out) {
  __shared__ _Float16 hbuf[16 * H_];
  const int blk = blockIdx.x, lane = threadIdx.x;
  const size_t row0 = (size_t)blk * 16;
  const int n = lane & 15, mh = (lane >> 4) * 8;
  v8f acc[8];
#pragma unroll
  for (int t = 0; t < 8; ++t) acc[t] = zero8();
#pragma unroll
  for (int kc = 0; kc < 4; ++kc) {
    v16h a = afrag_h(propH + row0 * H_ + kc * 32, H_);
#pragma unroll
    for (int t = 0; t < 8; ++t)
      acc[t] = wmma_f16(a, bfrag(ovw1T + t * 16 * H_ + kc * 32, H_), acc[t]);
  }
#pragma unroll
  for (int t = 0; t < 8; ++t) {
    const float bb = ov_b1[t * 16 + n];
#pragma unroll
    for (int j = 0; j < 8; ++j)
      hbuf[(mh + j) * H_ + t * 16 + n] = (_Float16)fmaxf(acc[t][j] + bb, 0.f);
  }
  __syncthreads();
  for (int idx = lane; idx < 48; idx += 32) {
    const int m = idx / 3, o = idx % 3;
    float s = ov_b2[o];
    for (int k = 0; k < H_; ++k) s += (float)hbuf[m * H_ + k] * ov_w2[k * 3 + o];
    out[(row0 + m) * 3 + o] = s;
  }
}

// ---------------------------------------------------------------------------
extern "C" void kernel_launch(void* const* d_in, const int* in_sizes, int n_in,
                              void* d_out, int out_size, void* d_ws, size_t ws_size,
                              hipStream_t stream) {
  (void)in_sizes; (void)n_in; (void)out_size; (void)ws_size;
  const float* rel_data = (const float*)d_in[0];
  const float* obj_data = (const float*)d_in[1];
  const float* sender   = (const float*)d_in[2];
  const float* receiver = (const float*)d_in[3];
  const float* rm_w1 = (const float*)d_in[4];  const float* rm_b1 = (const float*)d_in[5];
  const float* rm_w2 = (const float*)d_in[6];  const float* rm_b2 = (const float*)d_in[7];
  const float* rm_w3 = (const float*)d_in[8];  const float* rm_b3 = (const float*)d_in[9];
  const float* om_w1 = (const float*)d_in[10]; const float* om_b1 = (const float*)d_in[11];
  const float* om_w2 = (const float*)d_in[12]; const float* om_b2 = (const float*)d_in[13];
  const float* rmp_w = (const float*)d_in[14]; const float* rmp_b = (const float*)d_in[15];
  const float* omp_w = (const float*)d_in[16]; const float* omp_b = (const float*)d_in[17];
  const float* ov_w1 = (const float*)d_in[18]; const float* ov_b1 = (const float*)d_in[19];
  const float* ov_w2 = (const float*)d_in[20]; const float* ov_b2 = (const float*)d_in[21];

  char* ws = (char*)d_ws;
  size_t off = 0;
  auto alloc = [&](size_t bytes) -> void* {
    void* p = ws + off;
    off += (bytes + 255) & ~(size_t)255;
    return p;
  };
  _Float16* relEnc = (_Float16*)alloc((size_t)MREL * H_ * 2);
  _Float16* relOut = (_Float16*)alloc((size_t)MREL * H_ * 2);
  _Float16* objEnc = (_Float16*)alloc((size_t)MOBJ * H_ * 2);
  float*    propF  = (float*)   alloc((size_t)MOBJ * H_ * 4);
  _Float16* propH  = (_Float16*)alloc((size_t)MOBJ * H_ * 2);
  _Float16* propT  = (_Float16*)alloc((size_t)MOBJ * H_ * 2);
  float*    effect = (float*)   alloc((size_t)MOBJ * H_ * 4);
  _Float16* w2T    = (_Float16*)alloc((size_t)H_ * H_ * 2);
  _Float16* w3T    = (_Float16*)alloc((size_t)H_ * H_ * 2);
  _Float16* omw2T  = (_Float16*)alloc((size_t)H_ * H_ * 2);
  _Float16* ovw1T  = (_Float16*)alloc((size_t)H_ * H_ * 2);
  _Float16* rmpT   = (_Float16*)alloc((size_t)3 * H_ * H_ * 2);
  _Float16* ompT   = (_Float16*)alloc((size_t)3 * H_ * H_ * 2);

  prep_kernel<<<(MOBJ * H_) / 256, 256, 0, stream>>>(
      rm_w2, rm_w3, om_w2, ov_w1, rmp_w, omp_w,
      w2T, w3T, omw2T, ovw1T, rmpT, ompT, propF, propH, propT);

  encode_rel_kernel<<<MREL / 16, 32, 0, stream>>>(
      rel_data, rm_w1, rm_b1, w2T, rm_b2, w3T, rm_b3, relEnc);
  encode_obj_kernel<<<MOBJ / 16, 32, 0, stream>>>(
      obj_data, om_w1, om_b1, omw2T, om_b2, objEnc);

  for (int s = 0; s < 5; ++s) {
    rel_step_kernel<<<MREL / 16, 32, 0, stream>>>(
        sender, receiver, relEnc, propT, rmpT, rmp_b, relOut);
    effect_kernel<<<B_ * 4, 256, 0, stream>>>(receiver, relOut, effect);
    obj_step_kernel<<<MOBJ / 16, 32, 0, stream>>>(
        objEnc, effect, propF, propH, propT, ompT, omp_b);
  }

  out_head_kernel<<<MOBJ / 16, 32, 0, stream>>>(
      propH, ovw1T, ov_b1, ov_w2, ov_b2, (float*)d_out);
}
